// MultiModalMoE_5239860101489
// MI455X (gfx1250) — matrix operand order repaired and checked by reading, not compiled, not run on hardware
//
#include <hip/hip_runtime.h>

// Problem constants (match reference: B,S,D,E,K = 2,4096,1024,8,2)
#define BB 2
#define SS 4096
#define DD 1024
#define EE 8
#define KKTOP 2
#define NTOK (BB * SS)          // 8192 tokens
#define NPAIR (NTOK * KKTOP)    // 16384 (token, k) pairs
#define CAP NPAIR               // worst-case per-expert list capacity

// GEMM tiling
#define TM 32                   // rows (token-pairs) per workgroup
#define TN 128                  // output columns per workgroup (8 waves x 16)
#define TK 32                   // K step (bf16 WMMA K)
#define NC16 (DD / 16)          // 64 column-blocks of 16
#define NKB (DD / TK)           // 32 K-blocks of 32
#define MAXMT (NPAIR / TM + (EE - 1))   // 512 + 7 = 519 flat M-tiles max

typedef __attribute__((ext_vector_type(16))) __bf16          v16bf;
typedef __attribute__((ext_vector_type(16))) unsigned short  v16u;
typedef __attribute__((ext_vector_type(8)))  unsigned short  u16x8;
typedef __attribute__((ext_vector_type(4)))  unsigned short  u16x4;
typedef __attribute__((ext_vector_type(8)))  float           v8f;

// ---- workspace layout (bytes) ----
#define LISTS_OFF 256                                   // E*CAP ints = 512KB
#define Y_OFF   (LISTS_OFF + EE * CAP * 4)              // 524544; y: NPAIR*DD f32 = 64MB
#define XHI_OFF (Y_OFF   + NPAIR * DD * 4)              // x hi plane: NTOK*DD bf16 = 16MB
#define XLO_OFF (XHI_OFF + NTOK * DD * 2)               // x lo plane: 16MB
#define BH_OFF  (XLO_OFF + NTOK * DD * 2)               // W hi plane (swizzled): 16MB
#define BL_OFF  (BH_OFF  + EE * DD * DD * 2)            // W lo plane (swizzled): 16MB
// total ~= 128.6 MB

__device__ __forceinline__ unsigned short f32_to_bf16_rne(float f) {
    unsigned u = __float_as_uint(f);
    u += 0x7FFFu + ((u >> 16) & 1u);
    return (unsigned short)(u >> 16);
}

__device__ __forceinline__ v16bf concat_bf(u16x8 lo, u16x8 hi) {
    v16u t = __builtin_shufflevector(lo, hi, 0,1,2,3,4,5,6,7,8,9,10,11,12,13,14,15);
    return __builtin_bit_cast(v16bf, t);
}

__global__ __launch_bounds__(32) void moe_zero_counts(int* counts) {
    if (threadIdx.x < EE) counts[threadIdx.x] = 0;
}

__global__ __launch_bounds__(256) void moe_route(const int* __restrict__ topk,
                                                 int* __restrict__ counts,
                                                 int* __restrict__ lists) {
    int i = blockIdx.x * 256 + threadIdx.x;   // i = token*K + k
    if (i >= NPAIR) return;
    int e = topk[i];
    int pos = atomicAdd(&counts[e], 1);
    lists[e * CAP + pos] = i;
}

// Split x (f32, [tok][d]) into bf16 hi/lo planes, same layout. 4 elems/thread.
__global__ __launch_bounds__(256) void moe_split_x(const float* __restrict__ x,
                                                   unsigned short* __restrict__ xhi,
                                                   unsigned short* __restrict__ xlo) {
    int base = (blockIdx.x * 256 + threadIdx.x) * 4;
    if (base >= NTOK * DD) return;
    const float4 v = *(const float4*)(x + base);
    u16x4 h, l;
    float f[4] = {v.x, v.y, v.z, v.w};
    #pragma unroll
    for (int j = 0; j < 4; ++j) {
        unsigned short hb = f32_to_bf16_rne(f[j]);
        h[j] = hb;
        l[j] = f32_to_bf16_rne(f[j] - __uint_as_float(((unsigned)hb) << 16));
    }
    *(u16x4*)(xhi + base) = h;
    *(u16x4*)(xlo + base) = l;
}

// Split W and swizzle into per-lane WMMA B-fragment order:
//   plane[(((e*NC16 + c16)*NKB + kb)*32 + lane)*16 + el]
// where lane = h*16+ln holds column c16*16+ln, K = kb*32 + h*16 + el.
__global__ __launch_bounds__(256) void moe_swizzle_w(const float* __restrict__ W,
                                                     unsigned short* __restrict__ bh,
                                                     unsigned short* __restrict__ bl) {
    int t = blockIdx.x * 256 + threadIdx.x;          // one fragment-row (16 elems)
    if (t >= EE * NC16 * NKB * 32) return;
    int lane = t & 31;
    int r    = t >> 5;
    int kb   = r & (NKB - 1);
    r >>= 5;                                          // NKB == 32
    int c16  = r & (NC16 - 1);
    int e    = r >> 6;                                // NC16 == 64
    int ln = lane & 15, h = lane >> 4;
    int col = c16 * 16 + ln;

    const float* Wl = W + (size_t)e * DD * DD;
    v16u hv, lv;
    #pragma unroll
    for (int el = 0; el < 16; ++el) {
        int k = kb * TK + h * 16 + el;
        float f = Wl[(size_t)k * DD + col];
        unsigned short hb = f32_to_bf16_rne(f);
        hv[el] = hb;
        lv[el] = f32_to_bf16_rne(f - __uint_as_float(((unsigned)hb) << 16));
    }
    size_t off = (size_t)t * 16;
    *(v16u*)(bh + off) = hv;
    *(v16u*)(bl + off) = lv;
}

// Grouped GEMM: y[ent*D + col] = x[token] . W[e][:, col]  (unweighted)
// fp32-accurate via bf16 hi/lo split: a*b ~= ah*bh + ah*bl + al*bh
__global__ __launch_bounds__(256) void moe_grouped_gemm(
    const unsigned short* __restrict__ xhi, const unsigned short* __restrict__ xlo,
    const unsigned short* __restrict__ bhp, const unsigned short* __restrict__ blp,
    const int* __restrict__ counts, const int* __restrict__ lists,
    float* __restrict__ y)
{
    // Double-buffered A stage: [buf][plane][row][k], row stride 40
    // (80B rows: 16B-aligned, bank-spread). 2*2*32*40*2 = 10 KB LDS.
    __shared__ __align__(16) unsigned short As[2][2][TM][40];
    __shared__ int sEnt[TM];
    __shared__ int sTok[TM];

    const int tid = threadIdx.x;
    const int mt  = blockIdx.y;

    // map flat M-tile id -> (expert, row base) by scanning the 8 counts
    int e = 0, seen = 0, cnt = 0, rowBase = 0;
    for (; e < EE; ++e) {
        int c = counts[e];
        int t = (c + TM - 1) / TM;
        if (mt < seen + t) { rowBase = (mt - seen) * TM; cnt = c; break; }
        seen += t;
    }
    if (e == EE) return;   // uniform: past the last real tile

    if (tid < TM) {
        int r   = rowBase + tid;
        int ent = (r < cnt) ? lists[e * CAP + r] : 0;
        sEnt[tid] = ent;
        sTok[tid] = ent >> 1;   // K=2: token = ent / 2
    }
    __syncthreads();

    const int lane = tid & 31;
    const int wave = tid >> 5;
    const int h    = lane >> 4;   // K-half select
    const int ln   = lane & 15;

    // A staging role for this thread: plane p, row, 8-elem chunk
    const int sp    = tid >> 7;           // 0=hi, 1=lo
    const int sq    = tid & 127;
    const int srow  = sq >> 2;            // 0..31
    const int skoff = (sq & 3) * 8;       // 0,8,16,24
    const unsigned short* sxp = sp ? xlo : xhi;
    const size_t srcRow = (size_t)sTok[srow] * DD + skoff;

    // B fragment pointers: contiguous 16-elem chunk per lane, +512 elems per k-block
    const int c16 = blockIdx.x * (TN / 16) + wave;
    size_t bfrag = ((((size_t)e * NC16 + c16) * NKB) * 32 + lane) * 16;
    const unsigned short* bhq = bhp + bfrag;
    const unsigned short* blq = blp + bfrag;

    v8f acc0 = {};   // rows 0..15 of the M-tile
    v8f acc1 = {};   // rows 16..31

    // prologue: stage k-block 0 into buffer 0
    *(u16x8*)&As[0][sp][srow][skoff] = *(const u16x8*)(sxp + srcRow);

    int buf = 0;
    for (int kb = 0; kb < NKB; ++kb) {
        __syncthreads();   // stage for this kb visible; prior reads of buf^1 done

        // stage next k-block into the other buffer (overlaps with compute below)
        if (kb + 1 < NKB) {
            *(u16x8*)&As[buf ^ 1][sp][srow][skoff] =
                *(const u16x8*)(sxp + srcRow + (kb + 1) * TK);
        }

        // B fragments: 32B contiguous per lane (pre-swizzled, el -> K=h*16+el)
        v16bf Bh = __builtin_bit_cast(v16bf, *(const v16u*)bhq);
        v16bf Bl = __builtin_bit_cast(v16bf, *(const v16u*)blq);
        bhq += 32 * 16;
        blq += 32 * 16;

        // A fragments: lane row = ln (+16), K chunks [h*16,h*16+8) and [h*16+8,h*16+16)
        v16bf A0h = concat_bf(*(const u16x8*)&As[buf][0][ln][h * 16],
                              *(const u16x8*)&As[buf][0][ln][h * 16 + 8]);
        v16bf A0l = concat_bf(*(const u16x8*)&As[buf][1][ln][h * 16],
                              *(const u16x8*)&As[buf][1][ln][h * 16 + 8]);
        v16bf A1h = concat_bf(*(const u16x8*)&As[buf][0][16 + ln][h * 16],
                              *(const u16x8*)&As[buf][0][16 + ln][h * 16 + 8]);
        v16bf A1l = concat_bf(*(const u16x8*)&As[buf][1][16 + ln][h * 16],
                              *(const u16x8*)&As[buf][1][16 + ln][h * 16 + 8]);

        acc0 = __builtin_amdgcn_wmma_f32_16x16x32_bf16(false, A0h, false, Bl, (short)0, acc0, false, false);
        acc0 = __builtin_amdgcn_wmma_f32_16x16x32_bf16(false, A0l, false, Bh, (short)0, acc0, false, false);
        acc0 = __builtin_amdgcn_wmma_f32_16x16x32_bf16(false, A0h, false, Bh, (short)0, acc0, false, false);
        acc1 = __builtin_amdgcn_wmma_f32_16x16x32_bf16(false, A1h, false, Bl, (short)0, acc1, false, false);
        acc1 = __builtin_amdgcn_wmma_f32_16x16x32_bf16(false, A1l, false, Bh, (short)0, acc1, false, false);
        acc1 = __builtin_amdgcn_wmma_f32_16x16x32_bf16(false, A1h, false, Bh, (short)0, acc1, false, false);

        buf ^= 1;
    }

    // C/D layout: lanes 0-15 -> N=lane, M=v; lanes 16-31 -> N=lane-16, M=8+v
    const int col = blockIdx.x * TN + wave * 16 + ln;
    #pragma unroll
    for (int v = 0; v < 8; ++v) {
        int m0 = v + 8 * h;
        int r0 = rowBase + m0;
        if (r0 < cnt) y[(size_t)sEnt[m0] * DD + col] = acc0[v];
        int m1 = m0 + 16;
        int r1 = rowBase + m1;
        if (r1 < cnt) y[(size_t)sEnt[m1] * DD + col] = acc1[v];
    }
}

// out[t,d] = sum_k ew[t,k] * (y[(t*K+k), d] + bias[e_k, d])   (deterministic order)
__global__ __launch_bounds__(256) void moe_combine(
    const float* __restrict__ y, const float* __restrict__ bias,
    const int* __restrict__ topk, const float* __restrict__ ew,
    float* __restrict__ out)
{
    int gid  = blockIdx.x * 256 + threadIdx.x;
    int base = gid * 4;
    if (base >= NTOK * DD) return;
    int t = base / DD;
    int d = base % DD;

    int   e0 = topk[t * KKTOP + 0], e1 = topk[t * KKTOP + 1];
    float w0 = ew[t * KKTOP + 0],   w1 = ew[t * KKTOP + 1];

    const float4 y0 = *(const float4*)(y + (size_t)(t * KKTOP + 0) * DD + d);
    const float4 y1 = *(const float4*)(y + (size_t)(t * KKTOP + 1) * DD + d);
    const float4 b0 = *(const float4*)(bias + (size_t)e0 * DD + d);
    const float4 b1 = *(const float4*)(bias + (size_t)e1 * DD + d);

    float4 o;
    o.x = w0 * (y0.x + b0.x) + w1 * (y1.x + b1.x);
    o.y = w0 * (y0.y + b0.y) + w1 * (y1.y + b1.y);
    o.z = w0 * (y0.z + b0.z) + w1 * (y1.z + b1.z);
    o.w = w0 * (y0.w + b0.w) + w1 * (y1.w + b1.w);
    *(float4*)(out + (size_t)t * DD + d) = o;
}

extern "C" void kernel_launch(void* const* d_in, const int* in_sizes, int n_in,
                              void* d_out, int out_size, void* d_ws, size_t ws_size,
                              hipStream_t stream) {
    (void)in_sizes; (void)n_in; (void)out_size; (void)ws_size;

    const float* x    = (const float*)d_in[0];   // [B,S,D]
    const float* ew   = (const float*)d_in[1];   // [B,S,K]
    const int*   topk = (const int*)  d_in[2];   // [B,S,K]
    const float* W    = (const float*)d_in[3];   // [E,D,D]
    const float* bias = (const float*)d_in[4];   // [E,D]
    float* out = (float*)d_out;                  // [B,S,D]

    char* ws = (char*)d_ws;
    int*  counts = (int*)ws;
    int*  lists  = (int*)(ws + LISTS_OFF);
    float* y     = (float*)(ws + Y_OFF);
    unsigned short* xhi = (unsigned short*)(ws + XHI_OFF);
    unsigned short* xlo = (unsigned short*)(ws + XLO_OFF);
    unsigned short* bh  = (unsigned short*)(ws + BH_OFF);
    unsigned short* bl  = (unsigned short*)(ws + BL_OFF);

    moe_zero_counts<<<1, 32, 0, stream>>>(counts);
    moe_route<<<NPAIR / 256, 256, 0, stream>>>(topk, counts, lists);
    moe_split_x<<<(NTOK * DD / 4 + 255) / 256, 256, 0, stream>>>(x, xhi, xlo);
    moe_swizzle_w<<<(EE * NC16 * NKB * 32 + 255) / 256, 256, 0, stream>>>(W, bh, bl);

    dim3 ggrid(DD / TN, MAXMT);
    moe_grouped_gemm<<<ggrid, 256, 0, stream>>>(xhi, xlo, bh, bl, counts, lists, y);

    int nvec = (NTOK * DD) / 4;
    moe_combine<<<(nvec + 255) / 256, 256, 0, stream>>>(y, bias, topk, ew, out);
}
